// LinearWWeight_8770323218510
// MI455X (gfx1250) — matrix operand order repaired and checked by reading, not compiled, not run on hardware
//
#include <hip/hip_runtime.h>
#include <hip/hip_fp16.h>

#define N_  8
#define L_  1024
#define H_  12
#define D_  64
#define NH_ (N_*H_)

typedef __attribute__((ext_vector_type(16))) _Float16 v16h;
typedef __attribute__((ext_vector_type(8)))  float    v8f;
typedef __attribute__((ext_vector_type(4)))  float    f32x4;

#define WMMA_F16(a,b,c) __builtin_amdgcn_wmma_f32_16x16x32_f16(false,(a),false,(b),(short)0,(c),false,false)

// A-matrix fragment (16 rows x 32 k, f16). LDS tile stored row-major [row][k], stride in halves.
// Layout (CDNA5 ISA): k = 2p + 8*(p>=4) + 8*(lane>=16), row = lane&15.
__device__ __forceinline__ v16h frag_A(const _Float16* s, int row, int stride, int k0, int lane) {
    union { v16h v; unsigned u[8]; } r;
    const int ko = (lane >= 16) ? 8 : 0;
    const _Float16* base = s + row * stride;
#pragma unroll
    for (int p = 0; p < 8; ++p) {
        int k = k0 + 2*p + ((p >= 4) ? 8 : 0) + ko;
        r.u[p] = *(const unsigned*)(base + k);
    }
    return r.v;
}

// B-matrix fragment (32 k x 16 cols, f16). LDS tile stored [col][k].
// lane<16: col=lane, VGPR p holds k=2p+8*(p>=4),+1; lane>=16: k += 8.
__device__ __forceinline__ v16h frag_B(const _Float16* s, int col, int stride, int k0, int lane) {
    union { v16h v; unsigned u[8]; } r;
    const int ko = (lane >= 16) ? 8 : 0;
    const _Float16* base = s + col * stride;
#pragma unroll
    for (int p = 0; p < 8; ++p) {
        int k = k0 + 2*p + ((p >= 4) ? 8 : 0) + ko;
        r.u[p] = *(const unsigned*)(base + k);
    }
    return r.v;
}

// ---------------- 1. feature map: phi(x)=elu(x)+1, relayout [n,L,h,d] -> [n,h,L,d], f16 ----------
__global__ void phi_kernel(const float* __restrict__ q, const float* __restrict__ k,
                           const float* __restrict__ mask,
                           _Float16* __restrict__ Qh, _Float16* __restrict__ Kh) {
    const int NE = N_*L_*H_*D_;
    for (int i = blockIdx.x*blockDim.x + threadIdx.x; i < NE; i += gridDim.x*blockDim.x) {
        int dd = i & (D_-1);
        int t  = i >> 6;              // (n*L + l)*H + h
        int hh = t % H_;  t /= H_;
        int l  = t & (L_-1);
        int nn = t >> 10;
        size_t oi = ((size_t)((nn*H_ + hh)*L_ + l))*D_ + dd;
        float qv = __builtin_nontemporal_load(q + i);   // read-once streams: NT
        float kv = __builtin_nontemporal_load(k + i);
        float pq = qv > 0.f ? qv + 1.f : expf(qv);      // elu(x)+1
        float pk = kv > 0.f ? kv + 1.f : expf(kv);
        pk *= mask[nn*L_ + l];
        Qh[oi] = (_Float16)pq;
        Kh[oi] = (_Float16)pk;
    }
}

// ---------------- 3. Ksum[nh][d] = sum_s K[s][d] ----------------
__global__ void ksum_kernel(const _Float16* __restrict__ Kh, float* __restrict__ Ksum) {
    __shared__ float sm[256];
    int nh = blockIdx.x;
    int t = threadIdx.x;
    int d = t & 63, qq = t >> 6;
    const _Float16* kp = Kh + (size_t)nh * L_ * D_;
    float acc = 0.f;
    for (int s = qq; s < L_; s += 4) acc += (float)kp[s*D_ + d];
    sm[qq*64 + d] = acc;
    __syncthreads();
    if (t < 64) Ksum[nh*64 + t] = sm[t] + sm[64+t] + sm[128+t] + sm[192+t];
}

// ---------------- 4. KV[nh][m][d] = sum_s V[s,m]*K[s,d]  (WMMA, s-split x8, atomic reduce) -------
__global__ void kv_kernel(const float* __restrict__ V, const _Float16* __restrict__ Kh,
                          float* __restrict__ KV) {
    __shared__ _Float16 VT[64*34];   // A-tile: V^T[m][s'] for 32-wide s chunk
    __shared__ _Float16 KT[64*34];   // B-tile: K^T[d][s']
    int nh = blockIdx.y;
    int nn = nh / H_, hh = nh % H_;
    int s0 = blockIdx.x * 128;
    int t = threadIdx.x;
    int lane = t & 31, w = t >> 5;
    int mi = w >> 2, ni = w & 3;
    int lr = lane & 15;
    v8f acc1 = {}; v8f acc2 = {};
    for (int it = 0; it < 4; ++it) {
        int sb = s0 + it*32;
        {   // stage V (f32, [s][m], read-once -> NT) transposed into VT[m][s'], f16
            int sp = t >> 3, mseg = t & 7;
            const f32x4* vp = (const f32x4*)(V + ((size_t)((nn*L_ + sb + sp)*H_ + hh))*D_ + mseg*8);
            f32x4 a = __builtin_nontemporal_load(vp);
            f32x4 b = __builtin_nontemporal_load(vp + 1);
            int mb = mseg*8;
#pragma unroll
            for (int j = 0; j < 4; ++j) VT[(mb+j)*34 + sp]   = (_Float16)a[j];
#pragma unroll
            for (int j = 0; j < 4; ++j) VT[(mb+4+j)*34 + sp] = (_Float16)b[j];
        }
        {   // stage Kh ([s][d]) transposed into KT[d][s']
            int sp = t >> 3, seg = t & 7;
            union { uint4 u; _Float16 hv[8]; } kb;
            kb.u = *(const uint4*)(Kh + ((size_t)nh*L_ + sb + sp)*D_ + seg*8);
            int db = seg*8;
#pragma unroll
            for (int j = 0; j < 8; ++j) KT[(db+j)*34 + sp] = kb.hv[j];
        }
        __syncthreads();
        v16h a1 = frag_A(VT, mi*16 + lr, 34, 0, lane);
        v16h a2 = frag_A(VT, (mi+2)*16 + lr, 34, 0, lane);
        v16h b  = frag_B(KT, ni*16 + lr, 34, 0, lane);
        acc1 = WMMA_F16(a1, b, acc1);
        acc2 = WMMA_F16(a2, b, acc2);
        __syncthreads();
    }
    int moff = (lane >= 16) ? 8 : 0;
    int d = ni*16 + lr;
    float* kvp = KV + (size_t)nh*4096;
#pragma unroll
    for (int p = 0; p < 8; ++p) {
        __hip_atomic_fetch_add(&kvp[(mi*16 + p + moff)*64 + d], acc1[p],
                               __ATOMIC_RELAXED, __HIP_MEMORY_SCOPE_AGENT);
        __hip_atomic_fetch_add(&kvp[((mi+2)*16 + p + moff)*64 + d], acc2[p],
                               __ATOMIC_RELAXED, __HIP_MEMORY_SCOPE_AGENT);
    }
}

// ---------------- 5. per-row normalizers: Z = 1/(dot+eps), An = 1/(dot + L*eps) ------------------
__global__ void rowscale_kernel(const _Float16* __restrict__ Qh, const float* __restrict__ Ksum,
                                float* __restrict__ Z, float* __restrict__ An) {
    int r = blockIdx.x*blockDim.x + threadIdx.x;
    if (r >= NH_*L_) return;
    int nh = r >> 10;
    const _Float16* qp = Qh + (size_t)r * D_;
    const float* ks = Ksum + nh*64;
    float dot = 0.f;
#pragma unroll 8
    for (int d = 0; d < 64; ++d) dot += (float)qp[d] * ks[d];
    Z[r]  = 1.f / (dot + 1e-6f);
    An[r] = 1.f / (dot + (float)L_ * 1e-6f);
}

// ---------------- 6. attention_out[nh][l][s] = (Q K^T)[l][s] * An[l]  (403 MB writer) ------------
__global__ void attn_kernel(const _Float16* __restrict__ Qh, const _Float16* __restrict__ Kh,
                            const float* __restrict__ An, float* __restrict__ out) {
    __shared__ _Float16 Qs[32*72];
    __shared__ _Float16 Ks[64*72];
    __shared__ float Ans[32];
    int nh = blockIdx.z;
    int l0 = blockIdx.y * 32;
    int s0 = blockIdx.x * 64;
    int t = threadIdx.x;
    const _Float16* qp = Qh + ((size_t)nh*L_ + l0)*D_;
    for (int c = t; c < 512; c += 256) {           // 32x64 halves, 8B chunks
        int row = c >> 4, seg = c & 15;
        *(uint2*)&Qs[row*72 + seg*4] = *(const uint2*)(qp + row*64 + seg*4);
    }
    const _Float16* kp = Kh + ((size_t)nh*L_ + s0)*D_;
    for (int c = t; c < 1024; c += 256) {          // 64x64 halves
        int row = c >> 4, seg = c & 15;
        *(uint2*)&Ks[row*72 + seg*4] = *(const uint2*)(kp + row*64 + seg*4);
    }
    if (t < 32) Ans[t] = An[nh*L_ + l0 + t];
    __syncthreads();
    int lane = t & 31, w = t >> 5;
    int mi = w >> 2, ni = w & 3;                   // 2 x 4 tile grid of 16x16
    int lr = lane & 15;
    v8f acc = {};
    {
        v16h a0 = frag_A(Qs, mi*16 + lr, 72, 0, lane);
        v16h b0 = frag_B(Ks, ni*16 + lr, 72, 0, lane);
        acc = WMMA_F16(a0, b0, acc);
        v16h a1 = frag_A(Qs, mi*16 + lr, 72, 32, lane);
        v16h b1 = frag_B(Ks, ni*16 + lr, 72, 32, lane);
        acc = WMMA_F16(a1, b1, acc);
    }
    int moff = (lane >= 16) ? 8 : 0;
    int col = s0 + ni*16 + lr;
    float* op = out + (size_t)nh * L_ * L_;
#pragma unroll
    for (int p = 0; p < 8; ++p) {
        int lrow = mi*16 + p + moff;
        // streaming 403 MB output: non-temporal so Qh/Kh stay L2-resident
        __builtin_nontemporal_store(acc[p] * Ans[lrow], op + (size_t)(l0 + lrow)*L_ + col);
    }
}

// ---------------- 7. V[n,l,h,m] = Z[l] * sum_d Q[l,d]*KV[m,d] ------------------------------------
__global__ void vout_kernel(const _Float16* __restrict__ Qh, const float* __restrict__ KV,
                            const float* __restrict__ Z, float* __restrict__ Vo) {
    __shared__ _Float16 Qs[64*72];
    __shared__ _Float16 Bs[64*72];                 // Bs[col=m][k=d] = KV[m][d]
    __shared__ float Zs[64];
    int nh = blockIdx.y;
    int nn = nh / H_, hh = nh % H_;
    int l0 = blockIdx.x * 64;
    int t = threadIdx.x;
    const _Float16* qp = Qh + ((size_t)nh*L_ + l0)*D_;
    for (int c = t; c < 1024; c += 256) {
        int row = c >> 4, seg = c & 15;
        *(uint2*)&Qs[row*72 + seg*4] = *(const uint2*)(qp + row*64 + seg*4);
    }
    const float* kvp = KV + (size_t)nh*4096;
    for (int c = t; c < 1024; c += 256) {
        int row = c >> 4, seg = c & 15;
        f32x4 v4 = *(const f32x4*)(kvp + row*64 + seg*4);
        union { uint2 u; _Float16 hv[4]; } cv;
#pragma unroll
        for (int j = 0; j < 4; ++j) cv.hv[j] = (_Float16)v4[j];
        *(uint2*)&Bs[row*72 + seg*4] = cv.u;
    }
    if (t < 64) Zs[t] = Z[nh*L_ + l0 + t];
    __syncthreads();
    int lane = t & 31, w = t >> 5;
    int mi = w >> 2, ni = w & 3;                   // each wave: tiles (mi,ni) and (mi+2,ni)
    int lr = lane & 15;
    v8f acc1 = {}, acc2 = {};
#pragma unroll
    for (int kc = 0; kc < 2; ++kc) {
        int k0 = kc*32;
        v16h a1 = frag_A(Qs, mi*16 + lr, 72, k0, lane);
        v16h a2 = frag_A(Qs, (mi+2)*16 + lr, 72, k0, lane);
        v16h b  = frag_B(Bs, ni*16 + lr, 72, k0, lane);
        acc1 = WMMA_F16(a1, b, acc1);
        acc2 = WMMA_F16(a2, b, acc2);
    }
    int moff = (lane >= 16) ? 8 : 0;
    int m = ni*16 + lr;
#pragma unroll
    for (int p = 0; p < 8; ++p) {
        int r1 = mi*16 + p + moff;
        int r2 = (mi+2)*16 + p + moff;
        __builtin_nontemporal_store(acc1[p] * Zs[r1],
            Vo + ((size_t)((nn*L_ + l0 + r1)*H_ + hh))*D_ + m);
        __builtin_nontemporal_store(acc2[p] * Zs[r2],
            Vo + ((size_t)((nn*L_ + l0 + r2)*H_ + hh))*D_ + m);
    }
}

extern "C" void kernel_launch(void* const* d_in, const int* in_sizes, int n_in,
                              void* d_out, int out_size, void* d_ws, size_t ws_size,
                              hipStream_t stream) {
    const float* q    = (const float*)d_in[0];
    const float* k    = (const float*)d_in[1];
    const float* v    = (const float*)d_in[2];
    const float* mask = (const float*)d_in[3];
    float* out = (float*)d_out;

    char* ws = (char*)d_ws;
    size_t off = 0;
    const size_t NE = (size_t)N_*L_*H_*D_;
    _Float16* Qh = (_Float16*)(ws + off); off += NE*2;
    _Float16* Kh = (_Float16*)(ws + off); off += NE*2;
    float* Ksum = (float*)(ws + off); off += (size_t)NH_*64*4;
    float* KV   = (float*)(ws + off); off += (size_t)NH_*4096*4;
    float* Z    = (float*)(ws + off); off += (size_t)NH_*L_*4;
    float* An   = (float*)(ws + off); off += (size_t)NH_*L_*4;
    if (off > ws_size) return;   // ~27.6 MB workspace required

    phi_kernel<<<6144, 256, 0, stream>>>(q, k, mask, Qh, Kh);
    hipMemsetAsync(KV, 0, (size_t)NH_*4096*4, stream);
    ksum_kernel<<<NH_, 256, 0, stream>>>(Kh, Ksum);
    kv_kernel<<<dim3(8, NH_), 256, 0, stream>>>(v, Kh, KV);
    rowscale_kernel<<<(NH_*L_)/256, 256, 0, stream>>>(Qh, Ksum, Z, An);
    attn_kernel<<<dim3(L_/64, L_/32, NH_), 256, 0, stream>>>(Qh, Kh, An, out + NE);
    vout_kernel<<<dim3(L_/64, NH_), 256, 0, stream>>>(Qh, KV, Z, out);
}
